// GAT_90366111908391
// MI455X (gfx1250) — matrix operand order repaired and checked by reading, not compiled, not run on hardware
//
#include <hip/hip_runtime.h>
#include <hip/hip_bf16.h>

typedef __attribute__((ext_vector_type(16))) _Float16 v16h;
typedef __attribute__((ext_vector_type(8)))  float    v8f;
typedef __attribute__((ext_vector_type(4)))  int      v4i;

#define KDIM 128  // input dim of every GEMM in this net (128)

#if defined(__has_builtin)
#if __has_builtin(__builtin_amdgcn_global_load_async_to_lds_b128) && \
    __has_builtin(__builtin_amdgcn_s_wait_asynccnt)
#define USE_ASYNC_LDS 1
#endif
#endif
#ifndef USE_ASYNC_LDS
#define USE_ASYNC_LDS 0
#endif

#if USE_ASYNC_LDS
typedef __attribute__((address_space(1))) v4i* gv4i_p;  // global 16B chunk
typedef __attribute__((address_space(3))) v4i* lv4i_p;  // LDS 16B chunk
#endif

// ---------- order-preserving float<->uint map for atomic float max ----------
__device__ __forceinline__ unsigned ford(float f) {
  unsigned u = __float_as_uint(f);
  return (u & 0x80000000u) ? ~u : (u | 0x80000000u);
}
__device__ __forceinline__ float funord(unsigned u) {
  unsigned b = (u & 0x80000000u) ? (u ^ 0x80000000u) : ~u;
  return __uint_as_float(b);
}

// ---------------------------------------------------------------------------
// One-time per layer: convert W (f32, K x Mreal) into f16, pre-swizzled into
// this file's contiguous-K B-operand order (per 16x16x32 step):
//   lanes 0-15 hold K=kk*32+0..15 (elts 0..15), lanes 16-31 hold K=kk*32+16..31,
//   N = lane&15.  Flat index: ((n0*4+kk)*2+g)*256 + nl*16 + e.
// ---------------------------------------------------------------------------
__global__ void prep_w(const float* __restrict__ W, _Float16* __restrict__ Wswz,
                       int Mreal, int Mtiles)
{
  int i    = blockIdx.x * blockDim.x + threadIdx.x;
  int Mpad = Mtiles * 16;
  if (i >= KDIM * Mpad) return;
  int k = i / Mpad, n = i - k * Mpad;
  float w = (n < Mreal) ? W[k * Mreal + n] : 0.0f;
  int n0 = n >> 4, nl = n & 15, kk = k >> 5, k5 = k & 31;
  int g = (k5 >> 4) & 1, e = k5 & 15;
  Wswz[((n0 * 4 + kk) * 2 + g) * 256 + nl * 16 + e] = (_Float16)w;
}

// ---------------------------------------------------------------------------
// GEMM: H16[N x Mreal] = A[N x 128] * W[128 x Mreal], f16 WMMA, f32 accum,
// f16 output (halves edge-phase gather traffic; agg accumulation stays f32).
// Block = 256 threads = 8 waves; each wave owns a 16-row strip x (MT*16) cols.
// Pre-swizzled W is staged to LDS via global_load_async_to_lds_b128 (ASYNCcnt)
// when the toolchain exposes the builtin; else a plain b128 copy.
// A operand built in-register in the same contiguous-K order as B:
//   lane<16: elts 0..15 = K 0..15 (of the kk block)
//   lane>=16: elts 0..15 = K 16..31
// ---------------------------------------------------------------------------
template <int MT>
__global__ __launch_bounds__(256) void gemm_wmma_f16(
    const float* __restrict__ A, const _Float16* __restrict__ Wswz,
    _Float16* __restrict__ Hout, int N, int Mreal)
{
  __shared__ __align__(32) _Float16 Wh[MT * 2048];  // MT*4(kk)*2(group)*256 halves
  const int tid = threadIdx.x;

  // ---- stage pre-swizzled W into LDS: MT*256 chunks of 16B ----
#if USE_ASYNC_LDS
  for (int off = tid; off < MT * 256; off += 256) {
    __builtin_amdgcn_global_load_async_to_lds_b128(
        (gv4i_p)((v4i*)Wswz + off),
        (lv4i_p)((v4i*)&Wh[0] + off),
        0, 0);
  }
  __builtin_amdgcn_s_wait_asynccnt(0);
#else
  {
    const int4* src = (const int4*)Wswz;
    int4*       dst = (int4*)&Wh[0];
    for (int off = tid; off < MT * 256; off += 256) dst[off] = src[off];
  }
#endif
  __syncthreads();

  const int wave    = tid >> 5;
  const int lane    = tid & 31;
  const int rowBase = blockIdx.x * 128 + wave * 16;
  if (rowBase >= N) return;

  const int mrow   = lane & 15;
  const int hiHalf = lane >> 4;                 // which K sub-block this lane feeds
  int row  = rowBase + mrow;
  int rowc = row < N ? row : N - 1;             // clamp OOB rows (stores guarded)
  const float* arow = A + (size_t)rowc * KDIM;

  v8f c[MT];
  const v8f vzero = {0.f, 0.f, 0.f, 0.f, 0.f, 0.f, 0.f, 0.f};
#pragma unroll
  for (int t = 0; t < MT; ++t) c[t] = vzero;

#pragma unroll
  for (int kk = 0; kk < 4; ++kk) {
    const int kb = kk * 32 + hiHalf * 16;
    v16h a;
#pragma unroll
    for (int e = 0; e < 8; ++e) {
      a[e]     = (_Float16)arow[kb + e];
      a[e + 8] = (_Float16)arow[kb + 8 + e];
    }
#pragma unroll
    for (int t = 0; t < MT; ++t) {
      const v16h* bp = (const v16h*)&Wh[((t * 4 + kk) * 2 + hiHalf) * 256 + mrow * 16];
      v16h b = *bp;
      c[t] = __builtin_amdgcn_wmma_f32_16x16x32_f16(
          false, a, false, b, (short)0, c[t], false, false);
    }
  }

  // ---- store f16: C/D layout: vgpr j -> M = j + hiHalf*8, N = lane&15 ----
#pragma unroll
  for (int t = 0; t < MT; ++t) {
#pragma unroll
    for (int j = 0; j < 8; ++j) {
      int r   = rowBase + j + hiHalf * 8;
      int col = t * 16 + mrow;
      if (r < N && col < Mreal)
        Hout[(size_t)r * Mreal + col] = (_Float16)c[t][j];
    }
  }
}

// ---------------------------------------------------------------------------
// Per-node attention projections: es[n,h] = sum_c h[n,h,c]*a_s[h,c] (ed alike)
// ---------------------------------------------------------------------------
__global__ void node_alpha(const _Float16* __restrict__ Hf,
                           const float* __restrict__ a_s,
                           const float* __restrict__ a_d,
                           float* __restrict__ es, float* __restrict__ ed,
                           int N, int Hh, int C, int stride)
{
  int t = blockIdx.x * blockDim.x + threadIdx.x;
  if (t >= N * Hh) return;
  int n = t / Hh, hd = t - n * Hh;
  const _Float16* hp = Hf + (size_t)n * stride + hd * C;
  const float* asp = a_s + hd * C;
  const float* adp = a_d + hd * C;
  float s1 = 0.f, s2 = 0.f;
  for (int c = 0; c < C; ++c) {
    float v = (float)hp[c];
    s1 += v * asp[c];
    s2 += v * adp[c];
  }
  es[t] = s1; ed[t] = s2;
}

// ---------------------------------------------------------------------------
// Init: m = ord(-inf), s = 0, agg = 0
// ---------------------------------------------------------------------------
__global__ void init_bufs(unsigned* __restrict__ m, float* __restrict__ s,
                          float* __restrict__ agg, int NH, int NC)
{
  int i = blockIdx.x * blockDim.x + threadIdx.x;
  if (i < NH) { m[i] = ford(-INFINITY); s[i] = 0.f; }
  if (i < NC) agg[i] = 0.f;
}

// ---------------------------------------------------------------------------
// Edge pass 1: segment max of leakyrelu(es[src]+ed[dst]) over dst
// ---------------------------------------------------------------------------
__global__ void edge_max(const int* __restrict__ ei, int E, int Etot,
                         const float* __restrict__ es, const float* __restrict__ ed,
                         unsigned* __restrict__ m, int Hh)
{
  int e = blockIdx.x * blockDim.x + threadIdx.x;
  if (e >= Etot) return;
  int sN, dN;
  if (e < E) { sN = ei[e]; dN = ei[E + e]; } else { sN = dN = e - E; }
  for (int hd = 0; hd < Hh; ++hd) {
    float v = es[sN * Hh + hd] + ed[dN * Hh + hd];
    v = v > 0.f ? v : 0.2f * v;
    atomicMax(&m[dN * Hh + hd], ford(v));
  }
}

// ---------------------------------------------------------------------------
// Edge pass 2: segment sum of exp(e - m[dst]) over dst
// ---------------------------------------------------------------------------
__global__ void edge_sum(const int* __restrict__ ei, int E, int Etot,
                         const float* __restrict__ es, const float* __restrict__ ed,
                         const unsigned* __restrict__ m, float* __restrict__ ssum,
                         int Hh)
{
  int e = blockIdx.x * blockDim.x + threadIdx.x;
  if (e >= Etot) return;
  int sN, dN;
  if (e < E) { sN = ei[e]; dN = ei[E + e]; } else { sN = dN = e - E; }
  for (int hd = 0; hd < Hh; ++hd) {
    float v = es[sN * Hh + hd] + ed[dN * Hh + hd];
    v = v > 0.f ? v : 0.2f * v;
    atomicAdd(&ssum[dN * Hh + hd], __expf(v - funord(m[dN * Hh + hd])));
  }
}

// ---------------------------------------------------------------------------
// Edge pass 3: agg[dst] += alpha * h16[src]  — one wave32 per edge, coalesced.
// ---------------------------------------------------------------------------
__global__ __launch_bounds__(256) void edge_agg(
    const int* __restrict__ ei, int E, int Etot,
    const float* __restrict__ es, const float* __restrict__ ed,
    const unsigned* __restrict__ m, const float* __restrict__ ssum,
    const _Float16* __restrict__ Hf, float* __restrict__ agg,
    int Hh, int C, int Ctot)
{
  int wid  = (blockIdx.x * blockDim.x + threadIdx.x) >> 5;
  int lane = threadIdx.x & 31;
  if (wid >= Etot) return;
  int sN, dN;
  if (wid < E) { sN = ei[wid]; dN = ei[E + wid]; } else { sN = dN = wid - E; }

  // overlap the feature-row fetch with the dependent alpha scalar loads
  __builtin_prefetch(&Hf[(size_t)sN * Ctot + lane * 4], 0, 3);

  float alpha[4];
  for (int hd = 0; hd < Hh; ++hd) {
    float v = es[sN * Hh + hd] + ed[dN * Hh + hd];
    v = v > 0.f ? v : 0.2f * v;
    alpha[hd] = __expf(v - funord(m[dN * Hh + hd])) / ssum[dN * Hh + hd];
  }
  for (int c = lane; c < Ctot; c += 32) {
    int hd = c / C;
    atomicAdd(&agg[(size_t)dN * Ctot + c], alpha[hd] * (float)Hf[(size_t)sN * Ctot + c]);
  }
}

// ---------------------------------------------------------------------------
// Epilogue. mode 0: out = elu(agg+b); 1: out = elu(agg+b+out); 2: out = agg+b
// ---------------------------------------------------------------------------
__global__ void epilogue(const float* __restrict__ agg, const float* __restrict__ b,
                         float* __restrict__ out, int N, int Ctot, int mode)
{
  int i = blockIdx.x * blockDim.x + threadIdx.x;
  if (i >= N * Ctot) return;
  int col = i % Ctot;
  float v = agg[i] + b[col];
  if (mode == 1) v += out[i];
  if (mode != 2) v = v > 0.f ? v : (__expf(v) - 1.0f);
  out[i] = v;
}

// ---------------------------------------------------------------------------
extern "C" void kernel_launch(void* const* d_in, const int* in_sizes, int n_in,
                              void* d_out, int out_size, void* d_ws, size_t ws_size,
                              hipStream_t stream)
{
  const float* x  = (const float*)d_in[0];
  const int*   ei = (const int*)d_in[1];
  const float* Wl[4]  = {(const float*)d_in[2],  (const float*)d_in[6],
                         (const float*)d_in[10], (const float*)d_in[14]};
  const float* asl[4] = {(const float*)d_in[3],  (const float*)d_in[7],
                         (const float*)d_in[11], (const float*)d_in[15]};
  const float* adl[4] = {(const float*)d_in[4],  (const float*)d_in[8],
                         (const float*)d_in[12], (const float*)d_in[16]};
  const float* bl[4]  = {(const float*)d_in[5],  (const float*)d_in[9],
                         (const float*)d_in[13], (const float*)d_in[17]};

  const int N    = in_sizes[0] / 128;
  const int E    = in_sizes[1] / 2;
  const int Etot = E + N;   // + self loops

  // workspace layout
  _Float16* h16  = (_Float16*)d_ws;                   // N*128 halves
  float*    aggb = (float*)(h16 + (size_t)N * 128);   // N*128 f32
  float*    xcur = aggb + (size_t)N * 128;            // N*128 f32
  float*    es   = xcur + (size_t)N * 128;            // N*4
  float*    ed   = es   + (size_t)N * 4;              // N*4
  unsigned* mb   = (unsigned*)(ed + (size_t)N * 4);   // N*4
  float*    sb   = (float*)(mb + (size_t)N * 4);      // N*4
  _Float16* wswz = (_Float16*)(sb + (size_t)N * 4);   // 16384 halves (max MT=8)
  float*    outp = (float*)d_out;                     // N*40

  const int gemmGrid = (N + 127) / 128;
  const int edgeGrid = (Etot + 255) / 256;
  const int aggGrid  = (Etot + 7) / 8;                // one wave per edge

  // -------- hidden layers 0..2 (H=4, C=32, Ctot=128) --------
  const float* xin = x;
  for (int l = 0; l < 3; ++l) {
    prep_w<<<(KDIM * 128 + 255) / 256, 256, 0, stream>>>(Wl[l], wswz, 128, 8);
    gemm_wmma_f16<8><<<gemmGrid, 256, 0, stream>>>(xin, wswz, h16, N, 128);
    node_alpha<<<(N * 4 + 255) / 256, 256, 0, stream>>>(h16, asl[l], adl[l],
                                                        es, ed, N, 4, 32, 128);
    init_bufs<<<(N * 128 + 255) / 256, 256, 0, stream>>>(mb, sb, aggb,
                                                         N * 4, N * 128);
    edge_max<<<edgeGrid, 256, 0, stream>>>(ei, E, Etot, es, ed, mb, 4);
    edge_sum<<<edgeGrid, 256, 0, stream>>>(ei, E, Etot, es, ed, mb, sb, 4);
    edge_agg<<<aggGrid, 256, 0, stream>>>(ei, E, Etot, es, ed, mb, sb,
                                          h16, aggb, 4, 32, 128);
    epilogue<<<(N * 128 + 255) / 256, 256, 0, stream>>>(aggb, bl[l], xcur,
                                                        N, 128, l == 0 ? 0 : 1);
    xin = xcur;
  }

  // -------- output layer (H=1, C=40, Mtiles=3 padded to 48 cols) --------
  prep_w<<<(KDIM * 48 + 255) / 256, 256, 0, stream>>>(Wl[3], wswz, 40, 3);
  gemm_wmma_f16<3><<<gemmGrid, 256, 0, stream>>>(xcur, wswz, h16, N, 40);
  node_alpha<<<(N + 255) / 256, 256, 0, stream>>>(h16, asl[3], adl[3],
                                                  es, ed, N, 1, 40, 40);
  init_bufs<<<(N * 40 + 255) / 256, 256, 0, stream>>>(mb, sb, outp, N, N * 40);
  edge_max<<<edgeGrid, 256, 0, stream>>>(ei, E, Etot, es, ed, mb, 1);
  edge_sum<<<edgeGrid, 256, 0, stream>>>(ei, E, Etot, es, ed, mb, sb, 1);
  edge_agg<<<aggGrid, 256, 0, stream>>>(ei, E, Etot, es, ed, mb, sb,
                                        h16, outp, 1, 40, 40);
  epilogue<<<(N * 40 + 255) / 256, 256, 0, stream>>>(outp, bl[3], outp, N, 40, 2);
}
